// Efficient_SKAN_Base_16655883173891
// MI455X (gfx1250) — matrix operand order repaired and checked by reading, not compiled, not run on hardware
//
#include <hip/hip_runtime.h>
#include <cstdint>
#include <cstddef>

// Problem constants (from reference)
#define B_TOTAL 131072
#define IN_DIM  128
#define OUT_DIM 128
#define NUM_F   8
#define NGROUP  9          // 1 base (silu) group + 8 fourier groups
#define KTOT    (NGROUP * IN_DIM)   // 1152 virtual K

typedef __bf16 bf16_t;
typedef __attribute__((ext_vector_type(16))) __bf16 v16bf;
typedef __attribute__((ext_vector_type(8)))  __bf16 v8bf;
typedef __attribute__((ext_vector_type(8)))  float  v8f;
typedef __attribute__((ext_vector_type(4)))  int    v4i;

#define AS1 __attribute__((address_space(1)))
#define AS3 __attribute__((address_space(3)))

// ---- gfx1250 async global->LDS path (guarded; falls back to plain copies) ----
#if __has_builtin(__builtin_amdgcn_global_load_async_to_lds_b128)
#define USE_ASYNC 1
__device__ __forceinline__ void async_ld16(void* lds, const void* g) {
  // builtin signature (from clang diagnostic): (v4i32 AS1*, v4i32 AS3*, imm, imm)
  __builtin_amdgcn_global_load_async_to_lds_b128(
      (AS1 v4i*)(void*)g, (AS3 v4i*)lds, 0, 0);
}
#else
#define USE_ASYNC 0
#endif

#if __has_builtin(__builtin_amdgcn_s_wait_asynccnt)
#define WAIT_ASYNC() __builtin_amdgcn_s_wait_asynccnt(0)
#else
#define WAIT_ASYNC() asm volatile("s_wait_asynccnt 0x0" ::: "memory")
#endif

// -----------------------------------------------------------------------------
// Prep: fold coef & scale_sp into bf16 weights W'[9][OUT][IN]; reduce conv bias.
//   group 0      : base_weight[o][i]
//   group 1+f    : conv_w[f][o][i] * scale_sp[o] * coef[i][f]
//   bias[o]      : scale_sp[o] * sum_f conv_b[f][o]
// -----------------------------------------------------------------------------
__global__ __launch_bounds__(256) void skan_prep(
    const float* __restrict__ bw,   // [OUT][IN]
    const float* __restrict__ ssp,  // [OUT]
    const float* __restrict__ coef, // [IN][NUM_F]
    const float* __restrict__ cw,   // [NUM_F][OUT][IN]
    const float* __restrict__ cb,   // [NUM_F][OUT]
    bf16_t* __restrict__ wbf,       // [NGROUP][OUT][IN]
    float* __restrict__ bias)       // [OUT]
{
  int idx = blockIdx.x * 256 + threadIdx.x;
  if (idx < NGROUP * OUT_DIM * IN_DIM) {
    int g   = idx >> 14;          // /16384
    int rem = idx & 16383;
    int o   = rem >> 7;
    int i   = rem & 127;
    float w;
    if (g == 0) {
      w = bw[rem];
    } else {
      int f = g - 1;
      w = cw[f * (OUT_DIM * IN_DIM) + rem] * ssp[o] * coef[i * NUM_F + f];
    }
    wbf[idx] = (bf16_t)w;
  }
  if (idx < OUT_DIM) {
    float s = 0.f;
#pragma unroll
    for (int f = 0; f < NUM_F; ++f) s += cb[f * OUT_DIM + idx];
    bias[idx] = s * ssp[idx];
  }
}

// -----------------------------------------------------------------------------
// Fused kernel: per block, 128 batch rows x 128 outputs, K = 1152 in 9 groups.
// 256 threads = 8 wave32: 4 M-partitions x 2 N-partitions, each wave a 32x64
// output patch (2x4 wmma tiles, 8x v8f accumulators).
// -----------------------------------------------------------------------------
__global__ __launch_bounds__(256) void skan_fused(
    const float*  __restrict__ x,     // [B][IN]
    const float*  __restrict__ grid,  // [NUM_F]
    const bf16_t* __restrict__ wbf,   // [NGROUP][OUT][IN]
    const float*  __restrict__ bias,  // [OUT]
    float*        __restrict__ out)   // [B][OUT]
{
  // Padded strides to avoid 64-bank alignment (136 bf16 = 272B, 132 f32 = 528B)
  __shared__ __align__(16) float  xs[128 * 132];
  __shared__ __align__(16) bf16_t a_tile[128 * 136];
  __shared__ __align__(16) bf16_t w_tile[128 * 136];

  const int tid  = threadIdx.x;
  const int lane = tid & 31;
  const int wv   = tid >> 5;
  const int row0 = blockIdx.x * 128;

  // ---- stage x tile (f32) into LDS once ----
  {
    const float* xg = x + (size_t)row0 * IN_DIM;
#if USE_ASYNC
    for (int i = tid; i < 128 * 32; i += 256) {
      int r = i >> 5, c4 = i & 31;
      async_ld16(&xs[r * 132 + c4 * 4], xg + r * IN_DIM + c4 * 4);
    }
    WAIT_ASYNC();
#else
    for (int i = tid; i < 128 * 32; i += 256) {
      int r = i >> 5, c4 = i & 31;
      *(float4*)&xs[r * 132 + c4 * 4] = *(const float4*)(xg + r * IN_DIM + c4 * 4);
    }
#endif
  }

  v8f acc[2][4];
#pragma unroll
  for (int mt = 0; mt < 2; ++mt)
#pragma unroll
    for (int nt = 0; nt < 4; ++nt) acc[mt][nt] = 0.f;

  const int mbase = (wv & 3) * 32;   // wave's M offset (2 tiles of 16)
  const int nbase = (wv >> 2) * 64;  // wave's N offset (4 tiles of 16)
  const int lr = lane & 15;          // row/col within a 16x16 tile
  const int lh = lane >> 4;          // lane half (K-chunk selector)
  const int cp = tid & 63;           // activation column-pair 0..63
  const int r0 = tid >> 6;           // activation row phase 0..3

  for (int g = 0; g < NGROUP; ++g) {
    __syncthreads();  // all waves finished consuming previous group's tiles

    // ---- stage this group's 128x128 bf16 weight slab into LDS ----
    const bf16_t* wsrc = wbf + g * (OUT_DIM * IN_DIM);
#if USE_ASYNC
    for (int i = tid; i < 2048; i += 256) {       // 2048 chunks of 16B
      int r = i >> 4, c8 = i & 15;
      async_ld16(&w_tile[r * 136 + c8 * 8], wsrc + r * IN_DIM + c8 * 8);
    }
#else
    for (int i = tid; i < 2048; i += 256) {
      int r = i >> 4, c8 = i & 15;
      *(v8bf*)&w_tile[r * 136 + c8 * 8] = *(const v8bf*)(wsrc + r * IN_DIM + c8 * 8);
    }
#endif

    // ---- compute activation tile (bf16) while the weight copy is in flight ----
    const float fr = (g > 0) ? grid[g - 1] : 0.f;
    uint32_t* apk = (uint32_t*)a_tile;            // stride 68 dwords per row
#pragma unroll 4
    for (int r = r0; r < 128; r += 4) {
      float2 xv = *(const float2*)&xs[r * 132 + 2 * cp];
      float a0, a1;
      if (g == 0) {  // SiLU
        a0 = xv.x / (1.f + __expf(-xv.x));
        a1 = xv.y / (1.f + __expf(-xv.y));
      } else {       // sin(grid_f * x); coef folded into weights
        a0 = __sinf(fr * xv.x);
        a1 = __sinf(fr * xv.y);
      }
      uint32_t p = (uint32_t)__builtin_bit_cast(unsigned short, (bf16_t)a0)
                 | ((uint32_t)__builtin_bit_cast(unsigned short, (bf16_t)a1) << 16);
      apk[r * 68 + cp] = p;
    }
#if USE_ASYNC
    WAIT_ASYNC();
#endif
    __syncthreads();

    // ---- WMMA over this group's K=128 in four K=32 steps ----
#pragma unroll
    for (int ks = 0; ks < 4; ++ks) {
      union Frag { v16bf v; v8bf h[2]; };
      Frag A[2], Bf[4];
#pragma unroll
      for (int mt = 0; mt < 2; ++mt) {
        // A 16x32 bf16: lane holds row M=lr, K chunks at lh*8 and 16+lh*8
        int row = mbase + mt * 16 + lr;
        int k0  = ks * 32 + lh * 8;
        A[mt].h[0] = *(const v8bf*)&a_tile[row * 136 + k0];
        A[mt].h[1] = *(const v8bf*)&a_tile[row * 136 + k0 + 16];
      }
#pragma unroll
      for (int nt = 0; nt < 4; ++nt) {
        // B 32x16 bf16: lane holds column N=lr, K chunks at lh*8 and 16+lh*8
        int o  = nbase + nt * 16 + lr;
        int k0 = ks * 32 + lh * 8;
        Bf[nt].h[0] = *(const v8bf*)&w_tile[o * 136 + k0];
        Bf[nt].h[1] = *(const v8bf*)&w_tile[o * 136 + k0 + 16];
      }
#pragma unroll
      for (int mt = 0; mt < 2; ++mt)
#pragma unroll
        for (int nt = 0; nt < 4; ++nt)
          acc[mt][nt] = __builtin_amdgcn_wmma_f32_16x16x32_bf16(
              false, A[mt].v, false, Bf[nt].v,
              (short)0, acc[mt][nt], false, false);
    }
  }

  // ---- epilogue: bias + store (C layout: lane col N=lr, rows M=v+8*lh) ----
#pragma unroll
  for (int nt = 0; nt < 4; ++nt) {
    int col  = nbase + nt * 16 + lr;
    float bv = bias[col];
#pragma unroll
    for (int mt = 0; mt < 2; ++mt) {
      size_t rowg = (size_t)row0 + mbase + mt * 16 + lh * 8;
      float* op = out + rowg * OUT_DIM + col;
#pragma unroll
      for (int v = 0; v < 8; ++v)
        op[(size_t)v * OUT_DIM] = acc[mt][nt][v] + bv;
    }
  }
}

// -----------------------------------------------------------------------------
extern "C" void kernel_launch(void* const* d_in, const int* in_sizes, int n_in,
                              void* d_out, int out_size, void* d_ws, size_t ws_size,
                              hipStream_t stream) {
  (void)in_sizes; (void)n_in; (void)out_size; (void)ws_size;
  const float* x    = (const float*)d_in[0];  // [B, IN]
  const float* grid = (const float*)d_in[1];  // [NUM_F]
  const float* bw   = (const float*)d_in[2];  // [OUT, IN]
  const float* ssp  = (const float*)d_in[3];  // [OUT]
  const float* coef = (const float*)d_in[4];  // [IN, NUM_F]
  const float* cw   = (const float*)d_in[5];  // [NUM_F, OUT, IN]
  const float* cb   = (const float*)d_in[6];  // [NUM_F, OUT]
  float* out = (float*)d_out;

  bf16_t* wbf = (bf16_t*)d_ws;                                    // 294912 B
  float*  bias = (float*)((char*)d_ws + NGROUP * OUT_DIM * IN_DIM * sizeof(bf16_t));

  int prep_elems  = NGROUP * OUT_DIM * IN_DIM;
  int prep_blocks = (prep_elems + 255) / 256;
  skan_prep<<<prep_blocks, 256, 0, stream>>>(bw, ssp, coef, cw, cb, wbf, bias);

  int gemm_blocks = B_TOTAL / 128;  // 1024
  skan_fused<<<gemm_blocks, 256, 0, stream>>>(x, grid, wbf, bias, out);
}
